// StyleBlock_27367531610757
// MI455X (gfx1250) — matrix operand order repaired and checked by reading, not compiled, not run on hardware
//
#include <hip/hip_runtime.h>
#include <hip/hip_bf16.h>

typedef _Float16 h8  __attribute__((ext_vector_type(8)));
typedef _Float16 h16 __attribute__((ext_vector_type(16)));
typedef float    f8  __attribute__((ext_vector_type(8)));
typedef float    f4  __attribute__((ext_vector_type(4)));

static constexpr int CIN   = 512;
static constexpr int COUT  = 512;
static constexpr int WD    = 512;
static constexpr int BATCH = 16;
static constexpr int HW    = 4096;     // 64*64
static constexpr int KTOT  = 4608;     // 9*512, GEMM-K order: k = (kh*3+kw)*512 + i
static constexpr long XTOT = (long)BATCH * CIN * HW;  // 33554432 elements
static constexpr float C_LIN  = 0.04419417382415922f;   // 1/sqrt(512)
static constexpr float C_CONV = 0.014731391274719738f;  // 1/sqrt(512*9)

// ---------------- Kernel 0: x (f32) -> x16 (f16), vectorized 8 elems/thread
__global__ void tof16_kernel(const float* __restrict__ x, _Float16* __restrict__ x16) {
    long t = (long)blockIdx.x * blockDim.x + threadIdx.x;
    if (t * 8 >= XTOT) return;
    f4 a = reinterpret_cast<const f4*>(x)[t * 2];
    f4 c = reinterpret_cast<const f4*>(x)[t * 2 + 1];
    h8 o;
    #pragma unroll
    for (int q = 0; q < 4; ++q) { o[q] = (_Float16)a[q]; o[q + 4] = (_Float16)c[q]; }
    reinterpret_cast<h8*>(x16)[t] = o;
}

// ---------------- Kernel 1: style linear  A[b][i] = (w[b] . w_style[i])*c + b_style[i]
__global__ void style_kernel(const float* __restrict__ w,
                             const float* __restrict__ w_style,
                             const float* __restrict__ b_style,
                             float* __restrict__ A) {
    int t = blockIdx.x * blockDim.x + threadIdx.x;
    if (t >= BATCH * CIN) return;
    int b = t >> 9, i = t & 511;
    const float* wr = w + (long)b * WD;
    const float* sr = w_style + (long)i * WD;
    float acc = 0.f;
    for (int d = 0; d < WD; ++d) acc += wr[d] * sr[d];
    A[t] = acc * C_LIN + b_style[i];
}

// ---------------- Kernel 2: modulate + demodulate, emit f16 weights in tap-major K order
// W16[b][o][kk*512 + i] = w_conv[o][i][kk] * c_conv * A[b][i] * demod(b,o)
__global__ void modweight_kernel(const float* __restrict__ w_conv,
                                 const float* __restrict__ A,
                                 _Float16* __restrict__ W16) {
    int gwid = (blockIdx.x * blockDim.x + threadIdx.x) >> 5;
    int lane = threadIdx.x & 31;
    if (gwid >= BATCH * COUT) return;
    int b = gwid >> 9, o = gwid & 511;
    const float* wr = w_conv + (long)o * KTOT;   // layout [i*9 + kk]
    const float* Ab = A + (long)b * CIN;

    float ss = 0.f;
    for (int k = lane; k < KTOT; k += 32) {
        int i = k & 511, kk = k >> 9;
        float m = wr[i * 9 + kk] * C_CONV * Ab[i];
        ss += m * m;
    }
    for (int off = 16; off > 0; off >>= 1) ss += __shfl_xor(ss, off, 32);
    float demod = rsqrtf(ss + 1e-8f);

    _Float16* dst = W16 + (long)gwid * KTOT;
    for (int k = lane; k < KTOT; k += 32) {
        int i = k & 511, kk = k >> 9;
        dst[k] = (_Float16)(wr[i * 9 + kk] * C_CONV * Ab[i] * demod);
    }
}

// ---------------- Kernel 3: implicit-GEMM modulated conv via WMMA f16 -> f32
// block: 512 thr (16 waves), tile 256 Cout x 128 spatial (two image rows);
// 144 K-steps of 32; uniform filter tap per step (tap-major K). Each wave:
// 4 M-subtiles x 2 N-subtiles = 8 WMMAs per step from 4 A frags + 2 B frags.
__global__ __launch_bounds__(512) void conv_kernel(
        const _Float16* __restrict__ x16, const _Float16* __restrict__ W16,
        const float* __restrict__ noise, const float* __restrict__ scale_noise,
        const float* __restrict__ bias, float* __restrict__ out) {
    __shared__ __align__(16) _Float16 ldsB[2][128 * 32];  // [n][k], 8KB per buffer

    const int h0    = blockIdx.x * 2;     // two image rows h0, h0+1
    const int oBase = blockIdx.y * 256;   // Cout tile
    const int b     = blockIdx.z;         // batch

    const int tid  = threadIdx.x;
    const int lane = tid & 31;
    const int wid  = tid >> 5;            // 0..15
    const int msub0 = (wid & 3) * 4;      // M subtiles {msub0..msub0+3} of 16
    const int nsub0 = (wid >> 2) * 2;     // N subtiles {nsub0, nsub0+1} of 8
    const int hi   = lane >> 4;           // half-wave select
    const int lmod = lane & 15;

    // staging coords: thread -> (n = spatial col 0..127, 8 consecutive channels)
    const int sn  = tid & 127;
    const int skk = (tid >> 7) * 8;       // {0,8,16,24}
    const int srow = h0 + (sn >> 6);      // image row for this n
    const int scol = sn & 63;             // w coordinate

    const _Float16* arow[4];
    #pragma unroll
    for (int m = 0; m < 4; ++m)
        arow[m] = W16 + ((long)(b * COUT + oBase + (msub0 + m) * 16 + lmod)) * KTOT;
    const _Float16* xb = x16 + (long)b * CIN * HW;

    f8 acc[4][2];
    #pragma unroll
    for (int m = 0; m < 4; ++m) { acc[m][0] = (f8){}; acc[m][1] = (f8){}; }

    #pragma unroll 2
    for (int ks = 0; ks < 144; ++ks) {
        _Float16* buf = ldsB[ks & 1];

        // ---- stage im2col B tile: uniform tap per K-step
        const int kk = ks >> 4;             // filter tap 0..8
        const int kh = kk / 3;
        const int kw = kk - kh * 3;
        const int hh = srow + kh - 1;
        const int ww = scol + kw - 1;
        const bool vh = (unsigned)hh < 64u;
        const bool vw = (unsigned)ww < 64u;
        const bool valid = vh && vw;
        const int hc = vh ? hh : 0;
        const int wcc = vw ? ww : 0;
        const _Float16* src =
            xb + (((long)((ks & 15) * 32 + skk)) << 12) + (hc << 6) + wcc;

        h8 stv;
        #pragma unroll
        for (int q = 0; q < 8; ++q) {
            _Float16 v = src[(long)q << 12];   // 2B d16 loads, immediate offsets
            stv[q] = valid ? v : (_Float16)0.f;
        }
        *reinterpret_cast<h8*>(&buf[sn * 32 + skk]) = stv;
        __syncthreads();

        // ---- A fragments (16x32 f16): two contiguous 8-half runs per lane
        const long kOff = (long)ks * 32;
        h16 af[4];
        #pragma unroll
        for (int m = 0; m < 4; ++m) {
            h8 al = *reinterpret_cast<const h8*>(arow[m] + kOff + hi * 16);
            h8 ah = *reinterpret_cast<const h8*>(arow[m] + kOff + hi * 16 + 8);
            #pragma unroll
            for (int q = 0; q < 8; ++q) { af[m][q] = al[q]; af[m][q + 8] = ah[q]; }
        }

        // ---- B fragments (32x16 f16): 16 contiguous K halves per lane from LDS
        const h16 b0 = *reinterpret_cast<const h16*>(
            &buf[(nsub0 * 16 + lmod) * 32 + hi * 16]);
        const h16 b1 = *reinterpret_cast<const h16*>(
            &buf[(nsub0 * 16 + 16 + lmod) * 32 + hi * 16]);

        #pragma unroll
        for (int m = 0; m < 4; ++m) {
            acc[m][0] = __builtin_amdgcn_wmma_f32_16x16x32_f16(
                false, af[m], false, b0, (short)0, acc[m][0], false, false);
            acc[m][1] = __builtin_amdgcn_wmma_f32_16x16x32_f16(
                false, af[m], false, b1, (short)0, acc[m][1], false, false);
        }
    }

    // ---- epilogue: + scale_noise*noise + bias, LeakyReLU(0.2)
    const float snv = scale_noise[0];
    const int wc0 = nsub0 * 16 + lmod;     // spatial col index 0..127
    const int wc1 = wc0 + 16;
    const int row0 = h0 + (wc0 >> 6), col0 = wc0 & 63;
    const int row1 = h0 + (wc1 >> 6), col1 = wc1 & 63;
    const float nz0 = snv * noise[(long)b * HW + row0 * 64 + col0];
    const float nz1 = snv * noise[(long)b * HW + row1 * 64 + col1];
    #pragma unroll
    for (int m = 0; m < 4; ++m) {
        #pragma unroll
        for (int r = 0; r < 8; ++r) {
            int o = oBase + (msub0 + m) * 16 + r + hi * 8;  // C-layout M = r + 8*(lane>=16)
            float bo = bias[o];
            long obase = ((long)(b * COUT + o)) * HW;
            float v0 = acc[m][0][r] + nz0 + bo;
            out[obase + row0 * 64 + col0] = v0 >= 0.f ? v0 : 0.2f * v0;
            float v1 = acc[m][1][r] + nz1 + bo;
            out[obase + row1 * 64 + col1] = v1 >= 0.f ? v1 : 0.2f * v1;
        }
    }
}

extern "C" void kernel_launch(void* const* d_in, const int* in_sizes, int n_in,
                              void* d_out, int out_size, void* d_ws, size_t ws_size,
                              hipStream_t stream) {
    const float* x           = (const float*)d_in[0];
    const float* w           = (const float*)d_in[1];
    const float* noise       = (const float*)d_in[2];
    const float* w_style     = (const float*)d_in[3];
    const float* b_style     = (const float*)d_in[4];
    const float* w_conv      = (const float*)d_in[5];
    const float* scale_noise = (const float*)d_in[6];
    const float* bias        = (const float*)d_in[7];
    float* out = (float*)d_out;

    // workspace: A (64KB pad) | W16 f16 weights (75.5MB) | x16 f16 copy (67.1MB)
    float*    A   = (float*)d_ws;
    _Float16* W16 = (_Float16*)((char*)d_ws + 65536);
    _Float16* x16 = (_Float16*)((char*)d_ws + 65536 + (size_t)BATCH * COUT * KTOT * 2);

    tof16_kernel<<<(int)(XTOT / 8 / 256), 256, 0, stream>>>(x, x16);
    style_kernel<<<(BATCH * CIN + 255) / 256, 256, 0, stream>>>(w, w_style, b_style, A);
    modweight_kernel<<<(BATCH * COUT + 7) / 8, 256, 0, stream>>>(w_conv, A, W16);
    conv_kernel<<<dim3(32, 2, 16), 512, 0, stream>>>(x16, W16, noise, scale_noise, bias, out);
}
